// LinearAttention_39487929319717
// MI455X (gfx1250) — matrix operand order repaired and checked by reading, not compiled, not run on hardware
//
#include <hip/hip_runtime.h>

typedef __attribute__((ext_vector_type(16))) _Float16 v16h;
typedef __attribute__((ext_vector_type(4)))  _Float16 v4h;
typedef __attribute__((ext_vector_type(8)))  float    v8f;
typedef __attribute__((ext_vector_type(4)))  float    v4f;

#define HEADS   8
#define BATCH   16
#define CDIM    256
#define NTOK    4096

__device__ __forceinline__ v8f wmma16(v16h a, v16h b, v8f c) {
  return __builtin_amdgcn_wmma_f32_16x16x32_f16(false, a, false, b, (short)0, c, false, false);
}

// A fragment: 16x32 f16, row-major [row][k], leading dim ld (contiguous k = 16h + s).
__device__ __forceinline__ v16h load_a16(const _Float16* base, int row0, int k0, int ld) {
  const int lane = threadIdx.x & 31;
  const _Float16* p = base + (row0 + (lane & 15)) * ld + k0 + ((lane & 16) ? 16 : 0);
  v16h a;
#pragma unroll
  for (int i = 0; i < 8; ++i) { a[i] = p[i]; a[i + 8] = p[i + 8]; }
  return a;
}

// B fragment: 32x16 f16, stored [k][n], leading dim ld.
__device__ __forceinline__ v16h load_b16(const _Float16* base, int k0, int col0, int ld) {
  const int lane = threadIdx.x & 31;
  const int col = col0 + (lane & 15);
  const int kb  = k0 + ((lane & 16) ? 16 : 0);
  v16h b;
#pragma unroll
  for (int i = 0; i < 16; ++i) b[i] = base[(kb + i) * ld + col];
  return b;
}

// B fragment from transposed storage [n][k].
__device__ __forceinline__ v16h load_bT16(const _Float16* base, int k0, int col0, int ld) {
  const int lane = threadIdx.x & 31;
  const int col = col0 + (lane & 15);
  const int kb  = k0 + ((lane & 16) ? 16 : 0);
  v16h b;
#pragma unroll
  for (int i = 0; i < 16; ++i) b[i] = base[col * ld + kb + i];
  return b;
}

__device__ __forceinline__ v4h cvt4(v4f v) {
  v4h h;
#pragma unroll
  for (int k = 0; k < 4; ++k) h[k] = (_Float16)v[k];
  return h;
}

// ---------------------------------------------------------------------------
// Kernel 1: per (batch, head) -> context^T[e][d] with online token-softmax.
// ---------------------------------------------------------------------------
__device__ __forceinline__ void load_xtile(v4f* rx, const float* __restrict__ x,
                                           const float* __restrict__ memory,
                                           int b, int j, int tid) {
  if (j < 64) {
    const float* src = x + ((size_t)b * CDIM) * (size_t)NTOK + j * 64;
#pragma unroll
    for (int i = 0; i < 16; ++i) {
      const int q = tid + i * 256;
      rx[i] = *(const v4f*)(src + (size_t)(q >> 4) * NTOK + (q & 15) * 4);
    }
  } else {
#pragma unroll
    for (int i = 0; i < 16; ++i) {
      const int q = tid + i * 256;
      rx[i] = *(const v4f*)(memory + (q >> 4) * 64 + (q & 15) * 4);
    }
  }
}

__global__ void __launch_bounds__(256, 1)
ctx_kernel(const float* __restrict__ x, const float* __restrict__ memory,
           const float* __restrict__ Wqkv, float* __restrict__ ctxT) {
  __shared__ _Float16 sW[128 * 256];   // K rows 0..63, V rows 64..127   64KB
  __shared__ _Float16 sX[256 * 64];    // x tile [c][tok]                32KB
  __shared__ float    sK[64 * 64];     // K tile f32                     16KB
  __shared__ _Float16 sP[64 * 64];     // exp(K - m)                      8KB
  __shared__ _Float16 sV[64 * 64];     // V tile [e][tok]                 8KB
  __shared__ float    sRed[256];
  __shared__ float    sM[64], sL[64], sA[64];

  const int tid  = threadIdx.x;
  const int wv   = tid >> 5;
  const int lane = tid & 31;
  const int hi8  = (lane & 16) ? 8 : 0;
  const int bh   = blockIdx.x;
  const int b    = bh >> 3, h = bh & 7;

  // stage K/V weight slab (vectorized, f32 -> f16)
#pragma unroll 4
  for (int i = 0; i < 32; ++i) {
    const int q = tid + i * 256;          // quad index over 128x256 elems
    const int r = q >> 6, c = (q & 63) * 4;
    const int grow = (r < 64) ? (512 + h * 64 + r) : (1024 + h * 64 + (r - 64));
    *(v4h*)(sW + r * 256 + c) = cvt4(*(const v4f*)(Wqkv + (size_t)grow * 256 + c));
  }
  if (tid < 64) { sM[tid] = -3.0e38f; sL[tid] = 0.0f; sA[tid] = 1.0f; }

  v8f cacc[2] = {};
  const int tA = wv * 2, tB = wv * 2 + 1;
  const int rRed = tid & 63, pRed = tid >> 6;   // reduction role: row, quarter

  v4f rx[16];
  load_xtile(rx, x, memory, b, 0, tid);

  for (int j = 0; j < 65; ++j) {
    // spill pipelined registers to LDS as f16
#pragma unroll
    for (int i = 0; i < 16; ++i) {
      const int q = tid + i * 256;
      *(v4h*)(sX + (q >> 4) * 64 + (q & 15) * 4) = cvt4(rx[i]);
    }
    __syncthreads();
    if (j < 64) load_xtile(rx, x, memory, b, j + 1, tid);   // overlap with compute

    // KV(128x64) = W(128x256) @ X(256x64); wave wv owns row tile wv
    v8f acc[4] = {};
#pragma unroll
    for (int kc = 0; kc < 8; ++kc) {
      v16h a = load_a16(sW, wv * 16, kc * 32, 256);
#pragma unroll
      for (int ct = 0; ct < 4; ++ct) {
        v16h bb = load_b16(sX, kc * 32, ct * 16, 64);
        acc[ct] = wmma16(a, bb, acc[ct]);
      }
    }
#pragma unroll
    for (int ct = 0; ct < 4; ++ct) {
      const int col = ct * 16 + (lane & 15);
      if (wv < 4) {
#pragma unroll
        for (int i = 0; i < 8; ++i) sK[(wv * 16 + i + hi8) * 64 + col] = acc[ct][i];
      } else {
#pragma unroll
        for (int i = 0; i < 8; ++i) sV[((wv - 4) * 16 + i + hi8) * 64 + col] = (_Float16)acc[ct][i];
      }
    }
    __syncthreads();

    // online softmax stats per K row (all 256 threads, 4-way partials)
    {
      float mx = -3.0e38f;
#pragma unroll
      for (int t = 0; t < 16; ++t) mx = fmaxf(mx, sK[rRed * 64 + pRed * 16 + t]);
      sRed[pRed * 64 + rRed] = mx;
    }
    __syncthreads();
    if (tid < 64) {
      const float mx = fmaxf(fmaxf(sRed[tid], sRed[64 + tid]),
                             fmaxf(sRed[128 + tid], sRed[192 + tid]));
      const float mn = fmaxf(sM[tid], mx);
      sA[tid] = __expf(sM[tid] - mn);
      sM[tid] = mn;
    }
    __syncthreads();
    {
      const float mn = sM[rRed];
      float s = 0.f;
#pragma unroll
      for (int t = 0; t < 16; ++t) {
        const int o = rRed * 64 + pRed * 16 + t;
        const float e = __expf(sK[o] - mn);
        sP[o] = (_Float16)e;
        s += e;
      }
      sRed[pRed * 64 + rRed] = s;
    }
    __syncthreads();
    if (tid < 64)
      sL[tid] = sL[tid] * sA[tid] + (sRed[tid] + sRed[64 + tid] + sRed[128 + tid] + sRed[192 + tid]);

    // C = C*alpha + P(64x64) @ V^T(64x64); wave wv owns 2 of 16 tiles
    {
      const int tm0 = tA >> 2, tn0 = tA & 3, tm1 = tB >> 2, tn1 = tB & 3;
#pragma unroll
      for (int i = 0; i < 8; ++i) cacc[0][i] *= sA[tm0 * 16 + i + hi8];
#pragma unroll
      for (int i = 0; i < 8; ++i) cacc[1][i] *= sA[tm1 * 16 + i + hi8];
#pragma unroll
      for (int kc = 0; kc < 2; ++kc) {
        v16h a0 = load_a16(sP, tm0 * 16, kc * 32, 64);
        v16h b0 = load_bT16(sV, kc * 32, tn0 * 16, 64);
        cacc[0] = wmma16(a0, b0, cacc[0]);
        v16h a1 = load_a16(sP, tm1 * 16, kc * 32, 64);
        v16h b1 = load_bT16(sV, kc * 32, tn1 * 16, 64);
        cacc[1] = wmma16(a1, b1, cacc[1]);
      }
    }
    __syncthreads();
  }

  // normalize by running sum; write context TRANSPOSED: ctxT[e][d]
  {
    const int tm0 = tA >> 2, tn0 = tA & 3, tm1 = tB >> 2, tn1 = tB & 3;
    const int c0 = tn0 * 16 + (lane & 15), c1 = tn1 * 16 + (lane & 15);
#pragma unroll
    for (int i = 0; i < 8; ++i) {
      const int r0 = tm0 * 16 + i + hi8, r1 = tm1 * 16 + i + hi8;
      ctxT[(size_t)bh * 4096 + c0 * 64 + r0] = cacc[0][i] / sL[r0];
      ctxT[(size_t)bh * 4096 + c1 * 64 + r1] = cacc[1][i] / sL[r1];
    }
  }
}

// ---------------------------------------------------------------------------
// Kernel 2: per 64-token tile: Q -> softmax(d) -> ctx^T@q -> W_out -> LN.
// ---------------------------------------------------------------------------
__global__ void __launch_bounds__(256, 1)
apply_kernel(const float* __restrict__ x, const float* __restrict__ Wqkv,
             const float* __restrict__ Wout, const float* __restrict__ bout,
             const float* __restrict__ lng, const float* __restrict__ lnb,
             const float* __restrict__ ctxT, float* __restrict__ out) {
  __shared__ __align__(16) char pool[112128];
  _Float16* sX   = (_Float16*)(pool + 0);        // 256x64  f16  32KB
  _Float16* sWa  = (_Float16*)(pool + 32768);    // Wq 64x256 / Wo 256x64  32KB
  float*    sQ   = (float*)(pool + 65536);       // 64x64 f32   16KB
  _Float16* sQs  = (_Float16*)(pool + 81920);    // 64x64 f16    8KB
  _Float16* sCt  = (_Float16*)(pool + 90112);    // ctx^T 64x64  8KB
  _Float16* sOh  = (_Float16*)(pool + 98304);    // out_h 64x64  8KB
  float*    sRed = (float*)(pool + 106496);      // 256 f32
  float*    sRd2 = (float*)(pool + 107520);      // 256 f32
  float*    sSt  = (float*)(pool + 108544);      // 64 f32 (max / mean)
  float*    sIv  = (float*)(pool + 108800);      // 64 f32 (inv)
  float*    sLg  = (float*)(pool + 109056);      // 256 f32
  float*    sLb  = (float*)(pool + 110080);      // 256 f32
  float*    sBo  = (float*)(pool + 111104);      // 256 f32
  float*    sY   = (float*)(pool + 0);           // overlay 256x64 f32 64KB

  const int tid  = threadIdx.x;
  const int wv   = tid >> 5;
  const int lane = tid & 31;
  const int hi8  = (lane & 16) ? 8 : 0;
  const int blk  = blockIdx.x;
  const int b    = blk >> 6, tile = blk & 63;
  const int t0   = tile * 64;
  const int cRed = tid & 63, pRed = tid >> 6;

  // stage x tile + LN/bias params
  {
    const float* src = x + ((size_t)b * CDIM) * (size_t)NTOK + t0;
#pragma unroll
    for (int i = 0; i < 16; ++i) {
      const int q = tid + i * 256;
      const int c = q >> 4, tt = (q & 15) * 4;
      *(v4h*)(sX + c * 64 + tt) = cvt4(*(const v4f*)(src + (size_t)c * NTOK + tt));
    }
    sBo[tid] = bout[tid];
    sLg[tid] = lng[tid];
    sLb[tid] = lnb[tid];
  }

  v8f yacc[8] = {};                  // wave wv: rows wv*32..+32, 4 col tiles
  const int rt = wv & 3;             // Q/out row tile (0..3)
  const int cg = (wv >> 2) * 2;      // Q/out col tile base (0 or 2)

  for (int h = 0; h < 8; ++h) {
    // stage W_q head slab [r][c] and ctx^T [e][d] (both vectorized)
#pragma unroll
    for (int i = 0; i < 16; ++i) {
      const int q = tid + i * 256;                 // 4096 quads over 64x256
      const int r = q >> 6, c = (q & 63) * 4;
      *(v4h*)(sWa + r * 256 + c) = cvt4(*(const v4f*)(Wqkv + (size_t)(h * 64 + r) * 256 + c));
    }
#pragma unroll
    for (int i = 0; i < 4; ++i) {
      const int q = tid + i * 256;                 // 1024 quads over 64x64
      const int e = q >> 4, d = (q & 15) * 4;
      *(v4h*)(sCt + e * 64 + d) =
          cvt4(*(const v4f*)(ctxT + ((size_t)(b * 8 + h)) * 4096 + e * 64 + d));
    }
    __syncthreads();

    // Q(64x64) = Wq_h(64x256) @ X(256x64)
    v8f q[2] = {};
#pragma unroll
    for (int kc = 0; kc < 8; ++kc) {
      v16h a = load_a16(sWa, rt * 16, kc * 32, 256);
#pragma unroll
      for (int ct = 0; ct < 2; ++ct) {
        v16h bb = load_b16(sX, kc * 32, (cg + ct) * 16, 64);
        q[ct] = wmma16(a, bb, q[ct]);
      }
    }
#pragma unroll
    for (int ct = 0; ct < 2; ++ct) {
      const int col = (cg + ct) * 16 + (lane & 15);
#pragma unroll
      for (int i = 0; i < 8; ++i) sQ[(rt * 16 + i + hi8) * 64 + col] = q[ct][i];
    }
    __syncthreads();

    // softmax over head-dim per column (parallel 4-way partials), * SCALE
    {
      float mx = -3.0e38f;
#pragma unroll
      for (int r = 0; r < 16; ++r) mx = fmaxf(mx, sQ[(pRed * 16 + r) * 64 + cRed]);
      sRed[pRed * 64 + cRed] = mx;
    }
    __syncthreads();
    if (tid < 64)
      sSt[tid] = fmaxf(fmaxf(sRed[tid], sRed[64 + tid]),
                       fmaxf(sRed[128 + tid], sRed[192 + tid]));
    __syncthreads();
    {
      const float mx = sSt[cRed];
      float s = 0.f;
#pragma unroll
      for (int r = 0; r < 16; ++r) s += __expf(sQ[(pRed * 16 + r) * 64 + cRed] - mx);
      sRed[pRed * 64 + cRed] = s;
    }
    __syncthreads();
    if (tid < 64)
      sIv[tid] = 0.125f / (sRed[tid] + sRed[64 + tid] + sRed[128 + tid] + sRed[192 + tid]);
    __syncthreads();
    {
      const float mx = sSt[cRed], inv = sIv[cRed];
#pragma unroll
      for (int r = 0; r < 16; ++r) {
        const int row = pRed * 16 + r;
        sQs[row * 64 + cRed] = (_Float16)(__expf(sQ[row * 64 + cRed] - mx) * inv);
      }
    }
    __syncthreads();

    // out_h(64x64) = ctx^T(64x64) @ q_sm(64x64)
    v8f o[2] = {};
#pragma unroll
    for (int kc = 0; kc < 2; ++kc) {
      v16h a = load_a16(sCt, rt * 16, kc * 32, 64);
#pragma unroll
      for (int ct = 0; ct < 2; ++ct) {
        v16h bb = load_b16(sQs, kc * 32, (cg + ct) * 16, 64);
        o[ct] = wmma16(a, bb, o[ct]);
      }
    }
#pragma unroll
    for (int ct = 0; ct < 2; ++ct) {
      const int col = (cg + ct) * 16 + (lane & 15);
#pragma unroll
      for (int i = 0; i < 8; ++i) sOh[(rt * 16 + i + hi8) * 64 + col] = (_Float16)o[ct][i];
    }
    __syncthreads();

    // stage W_out head columns as A layout [c][e] (reuses sWa)
#pragma unroll
    for (int i = 0; i < 16; ++i) {
      const int q = tid + i * 256;                 // 4096 quads over 256x64
      const int r = q >> 4, e = (q & 15) * 4;
      *(v4h*)(sWa + r * 64 + e) = cvt4(*(const v4f*)(Wout + (size_t)r * 512 + h * 64 + e));
    }
    __syncthreads();

    // y(256x64) += Wout_h(256x64) @ out_h(64x64)
#pragma unroll
    for (int kc = 0; kc < 2; ++kc) {
#pragma unroll
      for (int rr = 0; rr < 2; ++rr) {
        v16h a = load_a16(sWa, wv * 32 + rr * 16, kc * 32, 64);
#pragma unroll
        for (int ct = 0; ct < 4; ++ct) {
          v16h bb = load_b16(sOh, kc * 32, ct * 16, 64);
          yacc[rr * 4 + ct] = wmma16(a, bb, yacc[rr * 4 + ct]);
        }
      }
    }
    __syncthreads();
  }

  // +bias, spill y to LDS overlay for the cross-wave LayerNorm
#pragma unroll
  for (int rr = 0; rr < 2; ++rr)
#pragma unroll
    for (int ct = 0; ct < 4; ++ct) {
      const int col = ct * 16 + (lane & 15);
#pragma unroll
      for (int i = 0; i < 8; ++i) {
        const int row = wv * 32 + rr * 16 + i + hi8;
        sY[row * 64 + col] = yacc[rr * 4 + ct][i] + sBo[row];
      }
    }
  __syncthreads();

  // channel LayerNorm per token column (parallel partials), store
  {
    float s = 0.f, s2 = 0.f;
#pragma unroll
    for (int r = 0; r < 64; ++r) {
      const float v = sY[(pRed * 64 + r) * 64 + cRed];
      s += v; s2 += v * v;
    }
    sRed[pRed * 64 + cRed] = s;
    sRd2[pRed * 64 + cRed] = s2;
  }
  __syncthreads();
  if (tid < 64) {
    const float s  = sRed[tid] + sRed[64 + tid] + sRed[128 + tid] + sRed[192 + tid];
    const float s2 = sRd2[tid] + sRd2[64 + tid] + sRd2[128 + tid] + sRd2[192 + tid];
    const float mean = s * (1.f / 256.f);
    const float var  = s2 * (1.f / 256.f) - mean * mean;
    sSt[tid] = mean;
    sIv[tid] = rsqrtf(var + 1e-5f);
  }
  __syncthreads();
  {
    const float mean = sSt[cRed], inv = sIv[cRed];
#pragma unroll
    for (int r = 0; r < 64; ++r) {
      const int row = pRed * 64 + r;
      out[((size_t)b * CDIM + row) * (size_t)NTOK + t0 + cRed] =
          (sY[row * 64 + cRed] - mean) * inv * sLg[row] + sLb[row];
    }
  }
}

extern "C" void kernel_launch(void* const* d_in, const int* in_sizes, int n_in,
                              void* d_out, int out_size, void* d_ws, size_t ws_size,
                              hipStream_t stream) {
  const float* x      = (const float*)d_in[0];
  const float* memory = (const float*)d_in[1];
  const float* Wqkv   = (const float*)d_in[2];
  const float* Wout   = (const float*)d_in[3];
  const float* bout   = (const float*)d_in[4];
  const float* lng    = (const float*)d_in[5];
  const float* lnb    = (const float*)d_in[6];
  float* out  = (float*)d_out;
  float* ctxT = (float*)d_ws;   // 128 heads * 64 * 64 floats = 2 MB

  ctx_kernel<<<dim3(BATCH * HEADS), dim3(256), 0, stream>>>(x, memory, Wqkv, ctxT);
  apply_kernel<<<dim3(BATCH * 64), dim3(256), 0, stream>>>(x, Wqkv, Wout, bout, lng, lnb, ctxT, out);
}
